// GatedGCNNet_89215060672864
// MI455X (gfx1250) — matrix operand order repaired and checked by reading, not compiled, not run on hardware
//
#include <hip/hip_runtime.h>
#include <hip/hip_bf16.h>

typedef __attribute__((ext_vector_type(8)))  float  v8f;
typedef __attribute__((ext_vector_type(16))) __bf16 v16bf;

#define NN   30000
#define NEDG 600000
#define HID  128

// ---------------- workspace layout (float offsets) ----------------
constexpr size_t OFF_H     = 0;                            // [NN,128] f32 node features
constexpr size_t OFF_HNEW  = OFF_H     + (size_t)NN*128;   // [NN,128] f32 pre-BN h_new
constexpr size_t OFF_ABDE  = OFF_HNEW  + (size_t)NN*128;   // [NN,512] f32 Ah|Bh|Dh|Eh
constexpr size_t OFF_NUM   = OFF_ABDE  + (size_t)NN*512;   // [NN,128]
constexpr size_t OFF_DEN   = OFF_NUM   + (size_t)NN*128;   // [NN,128]
constexpr size_t OFF_STATS = OFF_DEN   + (size_t)NN*128;   // 1024 floats
constexpr size_t OFF_E     = OFF_STATS + 1024;             // [NE,128] f32 e (later x2)
constexpr size_t OFF_ENEW  = OFF_E     + (size_t)NEDG*128; // [NE,128] f32 Ce/e_new
constexpr size_t OFF_PACKF = OFF_ENEW  + (size_t)NEDG*128; // packed bf16 weights (ushort)
constexpr size_t PACK_F    = 200000;                       // > 368640 ushorts
constexpr size_t OFF_HB    = OFF_PACKF + PACK_F;           // bf16 mirror of h  [NN*128 u16]
constexpr size_t OFF_EB    = OFF_HB + (size_t)NN*64;       // bf16 mirror of e / x1 [NE*128 u16]

constexpr size_t PK_SZ = 128*128;         // packed elems per HxH weight
constexpr size_t PK_W0 = 20*PK_SZ;        // mlp_W0 [256,128]
constexpr size_t PK_W1 = PK_W0 + 256*128; // mlp_W1 [128,64]

// ---------------- helpers ----------------
__device__ __forceinline__ unsigned short f2bf(float f) {
  unsigned u = __float_as_uint(f);
  unsigned r = u + 0x7FFFu + ((u >> 16) & 1u);   // RNE
  return (unsigned short)(r >> 16);
}

union BF16x16 { unsigned short s[16]; uint4 q[2]; v16bf v; };

// A fragment 16x32 from row-major bf16 activations: two 16B loads, no cvt
__device__ __forceinline__ v16bf load_a_bf(const unsigned short* A, long long lda,
                                           int m0, int k0, int lane) {
  const unsigned short* row = A + (size_t)(m0 + (lane & 15)) * lda + k0 + ((lane >> 4) * 8);
  BF16x16 u;
  u.q[0] = *(const uint4*)row;          // K = kb .. kb+7
  u.q[1] = *(const uint4*)(row + 16);   // K = kb+16 .. kb+23
  return u.v;
}

// B fragment 32x16 from LDS-staged packed panel: two ds_load_b128 per lane
__device__ __forceinline__ v16bf load_b_lds(const unsigned short* Bl, int ktiles,
                                            int kt, int nt, int lane) {
  const uint4* p = (const uint4*)(Bl + (((size_t)(nt * ktiles + kt)) << 9) + lane * 16);
  BF16x16 u; u.q[0] = p[0]; u.q[1] = p[1];
  return u.v;
}

// ---------------- small kernels ----------------
__global__ void zero_f32(float* p, long long n) {
  long long t = (long long)blockIdx.x * blockDim.x + threadIdx.x;
  if (t < n) p[t] = 0.f;
}

__global__ void pack_weight(const float* __restrict__ W, unsigned short* __restrict__ out,
                            int K, int Nc) {
  int t = blockIdx.x * blockDim.x + threadIdx.x;
  if (t >= K * Nc) return;
  int i    = t & 15;
  int lane = (t >> 4) & 31;
  int tile = t >> 9;
  int ktiles = K >> 5;
  int kt = tile % ktiles, nt = tile / ktiles;
  int col  = nt * 16 + (lane & 15);
  int krow = kt * 32 + ((lane >> 4) * 8) + (i & 7) + ((i >> 3) << 4);
  out[t] = f2bf(W[(size_t)krow * Nc + col]);
}

__global__ void embed_h(const float* __restrict__ h2, const float* __restrict__ W,
                        const float* __restrict__ b, float* __restrict__ out,
                        unsigned short* __restrict__ outb) {
  long long t = (long long)blockIdx.x * blockDim.x + threadIdx.x;
  if (t >= (long long)NN * 128) return;
  long long i = t >> 7; int c = (int)(t & 127);
  float v = h2[i * 2] * W[c] + h2[i * 2 + 1] * W[128 + c] + b[c];
  out[t] = v; outb[t] = f2bf(v);
}

__global__ void embed_e(const float* __restrict__ e1, const float* __restrict__ W,
                        const float* __restrict__ b, float* __restrict__ out,
                        unsigned short* __restrict__ outb) {
  long long t = (long long)blockIdx.x * blockDim.x + threadIdx.x;
  if (t >= (long long)NEDG * 128) return;
  long long i = t >> 7; int c = (int)(t & 127);
  float v = e1[i] * W[c] + b[c];
  out[t] = v; outb[t] = f2bf(v);
}

// C[M, NSUB*16] = A_bf16[M,K] @ Bpacked + bias
// B panel staged in LDS; each wave grid-strides over 16-row M tiles.
// acc zero-init (rematerializable, no spills); bias added in epilogue.
template<int NSUB, bool RELU>
__global__ void gemm_abf16(const unsigned short* __restrict__ A, long long lda,
                           const unsigned short* __restrict__ Bp,
                           const float* __restrict__ bias,
                           float* __restrict__ C, long long ldc,
                           int mtiles, int ktiles) {
  extern __shared__ unsigned short lbs[];
  const int nvec = (ktiles * NSUB * 512) / 8;          // uint4 count of panel
  for (int t = threadIdx.x; t < nvec; t += blockDim.x)
    ((uint4*)lbs)[t] = ((const uint4*)Bp)[t];
  __syncthreads();

  int lane  = threadIdx.x & 31;
  int wid   = (blockIdx.x * blockDim.x + threadIdx.x) >> 5;
  int nwave = (gridDim.x * blockDim.x) >> 5;

#pragma clang loop unroll(disable)
  for (int mt = wid; mt < mtiles; mt += nwave) {       // wave-uniform loop
    asm volatile("" ::: "memory");   // block LICM of LDS B-fragments (reg-pressure guard)
    int m0 = mt * 16;
    v8f acc[NSUB];
#pragma unroll
    for (int j = 0; j < NSUB; ++j)
#pragma unroll
      for (int r = 0; r < 8; ++r) acc[j][r] = 0.f;
    for (int kt = 0; kt < ktiles; ++kt) {
      v16bf a = load_a_bf(A, lda, m0, kt * 32, lane);
#pragma unroll
      for (int j = 0; j < NSUB; ++j) {
        v16bf b = load_b_lds(lbs, ktiles, kt, j, lane);
        acc[j] = __builtin_amdgcn_wmma_f32_16x16x32_bf16(false, a, false, b,
                                                         (short)0, acc[j], false, false);
      }
    }
#pragma unroll
    for (int j = 0; j < NSUB; ++j) {
      int col = j * 16 + (lane & 15);
      float bv = bias[col];
#pragma unroll
      for (int r = 0; r < 8; ++r) {
        int row = m0 + r + ((lane >> 4) * 8);
        float v = acc[j][r] + bv;
        if (RELU) v = fmaxf(v, 0.f);
        C[(size_t)row * ldc + col] = v;
      }
    }
  }
}

// readout GEMM1: A rows = concat(h_bf[src[e]], h_bf[dst[e]]) [NE,256] -> relu -> bf16 x1
__global__ void gemm_gather_bf16(const unsigned short* __restrict__ hb,
                                 const int* __restrict__ src, const int* __restrict__ dst,
                                 const unsigned short* __restrict__ Bp,
                                 const float* __restrict__ bias,
                                 unsigned short* __restrict__ Cb, int mtiles) {
  const int ktiles = 8;                                // K = 256
  extern __shared__ unsigned short lbs[];
  const int nvec = (ktiles * 8 * 512) / 8;             // 64KB panel
  for (int t = threadIdx.x; t < nvec; t += blockDim.x)
    ((uint4*)lbs)[t] = ((const uint4*)Bp)[t];
  __syncthreads();

  int lane  = threadIdx.x & 31;
  int wid   = (blockIdx.x * blockDim.x + threadIdx.x) >> 5;
  int nwave = (gridDim.x * blockDim.x) >> 5;

#pragma clang loop unroll(disable)
  for (int mt = wid; mt < mtiles; mt += nwave) {
    asm volatile("" ::: "memory");   // block LICM of LDS B-fragments
    int m0 = mt * 16;
    int e  = m0 + (lane & 15);
    const unsigned short* rs = hb + (size_t)src[e] * 128;
    const unsigned short* rd = hb + (size_t)dst[e] * 128;
    v8f acc[8];
#pragma unroll
    for (int j = 0; j < 8; ++j)
#pragma unroll
      for (int r = 0; r < 8; ++r) acc[j][r] = 0.f;
    for (int kt = 0; kt < ktiles; ++kt) {
      const unsigned short* row = (kt >= 4) ? rd : rs;
      int kb = (kt >= 4 ? kt * 32 - 128 : kt * 32) + ((lane >> 4) * 8);
      BF16x16 u;
      u.q[0] = *(const uint4*)(row + kb);
      u.q[1] = *(const uint4*)(row + kb + 16);
      v16bf a = u.v;
#pragma unroll
      for (int j = 0; j < 8; ++j) {
        v16bf b = load_b_lds(lbs, ktiles, kt, j, lane);
        acc[j] = __builtin_amdgcn_wmma_f32_16x16x32_bf16(false, a, false, b,
                                                         (short)0, acc[j], false, false);
      }
    }
#pragma unroll
    for (int j = 0; j < 8; ++j) {
      int col = j * 16 + (lane & 15);
      float bv = bias[col];
#pragma unroll
      for (int r = 0; r < 8; ++r) {
        int row = m0 + r + ((lane >> 4) * 8);
        Cb[(size_t)row * 128 + col] = f2bf(fmaxf(acc[j][r] + bv, 0.f));
      }
    }
  }
}

// wave-per-edge: e_new = Ce + Dh[src] + Eh[dst]; sig gate; atomic scatter num/den
__global__ void edge_msg(float* __restrict__ enew, const float* __restrict__ abde,
                         const int* __restrict__ src, const int* __restrict__ dst,
                         float* __restrict__ num, float* __restrict__ den) {
  int e = blockIdx.x * 8 + (threadIdx.x >> 5);
  if (e >= NEDG) return;
  int lane = threadIdx.x & 31;
  int f = lane * 4;
  int s = src[e], d = dst[e];
  float4 ce = *(float4*)(enew + (size_t)e * 128 + f);
  float4 dh = *(const float4*)(abde + (size_t)s * 512 + 256 + f);
  float4 eh = *(const float4*)(abde + (size_t)d * 512 + 384 + f);
  float4 bh = *(const float4*)(abde + (size_t)s * 512 + 128 + f);
  float4 en;
  en.x = ce.x + dh.x + eh.x; en.y = ce.y + dh.y + eh.y;
  en.z = ce.z + dh.z + eh.z; en.w = ce.w + dh.w + eh.w;
  *(float4*)(enew + (size_t)e * 128 + f) = en;
  float4 sg;
  sg.x = 1.f / (1.f + __expf(-en.x)); sg.y = 1.f / (1.f + __expf(-en.y));
  sg.z = 1.f / (1.f + __expf(-en.z)); sg.w = 1.f / (1.f + __expf(-en.w));
  float* np = num + (size_t)d * 128 + f;
  float* dp = den + (size_t)d * 128 + f;
  atomicAdd(np + 0, sg.x * bh.x); atomicAdd(np + 1, sg.y * bh.y);
  atomicAdd(np + 2, sg.z * bh.z); atomicAdd(np + 3, sg.w * bh.w);
  atomicAdd(dp + 0, sg.x); atomicAdd(dp + 1, sg.y);
  atomicAdd(dp + 2, sg.z); atomicAdd(dp + 3, sg.w);
}

__global__ void colstats(const float* __restrict__ X, long long rows,
                         float* __restrict__ sum, float* __restrict__ sumsq) {
  int c = threadIdx.x & 127;
  long long r0 = (long long)blockIdx.x * 2 + (threadIdx.x >> 7);
  long long step = (long long)gridDim.x * 2;
  float s = 0.f, q = 0.f;
  for (long long r = r0; r < rows; r += step) {
    float v = X[r * 128 + c];
    s += v; q += v * v;
  }
  atomicAdd(&sum[c], s); atomicAdd(&sumsq[c], q);
}

__global__ void bn_finalize(const float* __restrict__ sum, const float* __restrict__ sumsq,
                            const float* __restrict__ g, const float* __restrict__ b,
                            float rows, float* __restrict__ scale, float* __restrict__ shift) {
  int c = threadIdx.x;
  if (c >= 128) return;
  float m = sum[c] / rows;
  float v = sumsq[c] / rows - m * m;
  float rstd = rsqrtf(v + 1e-5f);
  scale[c] = g[c] * rstd;
  shift[c] = b[c] - g[c] * rstd * m;
}

__global__ void node_combine(const float* __restrict__ abde, const float* __restrict__ num,
                             const float* __restrict__ den, float* __restrict__ hnew) {
  long long t = (long long)blockIdx.x * blockDim.x + threadIdx.x;
  if (t >= (long long)NN * 128) return;
  long long i = t >> 7; int c = (int)(t & 127);
  hnew[t] = abde[i * 512 + c] + num[t] / (den[t] + 1e-6f);
}

// res += relu(x*scale+shift); also refresh bf16 mirror of res
__global__ void apply_bn_relu_res(float* __restrict__ res, unsigned short* __restrict__ mir,
                                  const float* __restrict__ x,
                                  const float* __restrict__ scale,
                                  const float* __restrict__ shift, long long n) {
  long long t = (long long)blockIdx.x * blockDim.x + threadIdx.x;
  if (t >= n) return;
  int c = (int)(t & 127);
  float v = res[t] + fmaxf(x[t] * scale[c] + shift[c], 0.f);
  res[t] = v; mir[t] = f2bf(v);
}

__global__ void final_mlp(const float* __restrict__ x2, const float* __restrict__ W2,
                          const float* __restrict__ b2, float* __restrict__ out) {
  __shared__ float w[128];
  __shared__ float bb[2];
  if (threadIdx.x < 128) w[threadIdx.x] = W2[threadIdx.x];
  if (threadIdx.x < 2)   bb[threadIdx.x] = b2[threadIdx.x];
  __syncthreads();
  long long e = (long long)blockIdx.x * blockDim.x + threadIdx.x;
  if (e >= NEDG) return;
  const float* r = x2 + e * 64;
  float a0 = bb[0], a1 = bb[1];
#pragma unroll
  for (int k = 0; k < 64; ++k) {
    float v = r[k];
    a0 += v * w[k * 2];
    a1 += v * w[k * 2 + 1];
  }
  out[e * 2]     = a0;
  out[e * 2 + 1] = a1;
}

// ---------------- driver ----------------
static inline int gemm_blocks(int mtiles) {
  int b = (mtiles + 7) / 8;
  return b < 512 ? b : 512;
}

extern "C" void kernel_launch(void* const* d_in, const int* in_sizes, int n_in,
                              void* d_out, int out_size, void* d_ws, size_t ws_size,
                              hipStream_t stream) {
  const float* in_h   = (const float*)d_in[0];
  const float* in_e   = (const float*)d_in[1];
  const int*   src    = (const int*)d_in[2];
  const int*   dst    = (const int*)d_in[3];
  const float* embhW  = (const float*)d_in[4];
  const float* embhB  = (const float*)d_in[5];
  const float* embeW  = (const float*)d_in[6];
  const float* embeB  = (const float*)d_in[7];
  const float* bnh_g  = (const float*)d_in[18];
  const float* bnh_b  = (const float*)d_in[19];
  const float* bne_g  = (const float*)d_in[20];
  const float* bne_b  = (const float*)d_in[21];
  const float* mlpW0  = (const float*)d_in[22];
  const float* mlpb0  = (const float*)d_in[23];
  const float* mlpW1  = (const float*)d_in[24];
  const float* mlpb1  = (const float*)d_in[25];
  const float* mlpW2  = (const float*)d_in[26];
  const float* mlpb2  = (const float*)d_in[27];

  float* ws = (float*)d_ws;
  float* B0 = ws + OFF_H;
  float* B6 = ws + OFF_HNEW;
  float* B3 = ws + OFF_ABDE;
  float* Bn = ws + OFF_NUM;
  float* Bd = ws + OFF_DEN;
  float* ST = ws + OFF_STATS;
  float* B1 = ws + OFF_E;       // e f32 (later x2 f32)
  float* B2 = ws + OFF_ENEW;    // Ce/e_new f32
  unsigned short* PK = (unsigned short*)(ws + OFF_PACKF);
  unsigned short* HB = (unsigned short*)(ws + OFF_HB);   // h bf16 mirror
  unsigned short* EB = (unsigned short*)(ws + OFF_EB);   // e bf16 mirror / x1 bf16

  float* sum = ST, * sumsq = ST + 128;
  float* esc = ST + 256, * esh = ST + 384;
  float* hsc = ST + 512, * hsh = ST + 640;

  // ---- pack all weights to bf16 B-fragments ----
  for (int l = 0; l < 4; ++l)
    for (int w = 0; w < 5; ++w) {
      const float* W = (const float*)d_in[8 + 2 * w] + (size_t)l * 128 * 128;
      pack_weight<<<(128 * 128) / 256, 256, 0, stream>>>(W, PK + ((size_t)l * 5 + w) * PK_SZ, 128, 128);
    }
  pack_weight<<<(256 * 128) / 256, 256, 0, stream>>>(mlpW0, PK + PK_W0, 256, 128);
  pack_weight<<<(128 * 64) / 256, 256, 0, stream>>>(mlpW1, PK + PK_W1, 128, 64);

  // ---- embeddings (f32 + bf16 mirrors) ----
  embed_h<<<(NN * 128) / 256, 256, 0, stream>>>(in_h, embhW, embhB, B0, HB);
  embed_e<<<((long long)NEDG * 128) / 256, 256, 0, stream>>>(in_e, embeW, embeB, B1, EB);

  const int NODE_MT = NN / 16;      // 1875
  const int EDGE_MT = NEDG / 16;    // 37500
  const int colofs[4] = {0, 128, 256, 384};   // A,B,D,E
  const int widx[4]   = {0, 1, 3, 4};

  for (int l = 0; l < 4; ++l) {
    zero_f32<<<(NN * 256 + 255) / 256, 256, 0, stream>>>(Bn, (long long)NN * 256);
    // node GEMMs A,B,D,E -> B3 column blocks (B panel in LDS, 32KB)
    for (int k = 0; k < 4; ++k) {
      int w = widx[k];
      const float* bias = (const float*)d_in[9 + 2 * w] + (size_t)l * 128;
      gemm_abf16<8, false><<<gemm_blocks(NODE_MT), 256, 32768, stream>>>(
          HB, 128, PK + ((size_t)l * 5 + w) * PK_SZ, bias,
          B3 + colofs[k], 512, NODE_MT, 4);
    }
    // edge GEMM: Ce -> B2
    {
      const float* bias = (const float*)d_in[13] + (size_t)l * 128;
      gemm_abf16<8, false><<<gemm_blocks(EDGE_MT), 256, 32768, stream>>>(
          EB, 128, PK + ((size_t)l * 5 + 2) * PK_SZ, bias,
          B2, 128, EDGE_MT, 4);
    }
    edge_msg<<<NEDG / 8, 256, 0, stream>>>(B2, B3, src, dst, Bn, Bd);
    // BN stats for e_new
    zero_f32<<<1, 256, 0, stream>>>(sum, 256);
    colstats<<<1024, 256, 0, stream>>>(B2, (long long)NEDG, sum, sumsq);
    bn_finalize<<<1, 128, 0, stream>>>(sum, sumsq, bne_g + l * 128, bne_b + l * 128,
                                       (float)NEDG, esc, esh);
    // node combine + BN stats for h_new
    node_combine<<<(NN * 128) / 256, 256, 0, stream>>>(B3, Bn, Bd, B6);
    zero_f32<<<1, 256, 0, stream>>>(sum, 256);
    colstats<<<1024, 256, 0, stream>>>(B6, (long long)NN, sum, sumsq);
    bn_finalize<<<1, 128, 0, stream>>>(sum, sumsq, bnh_g + l * 128, bnh_b + l * 128,
                                       (float)NN, hsc, hsh);
    // residual updates (+ refresh bf16 mirrors)
    apply_bn_relu_res<<<(NN * 128) / 256, 256, 0, stream>>>(B0, HB, B6, hsc, hsh,
                                                            (long long)NN * 128);
    apply_bn_relu_res<<<((long long)NEDG * 128) / 256, 256, 0, stream>>>(B1, EB, B2, esc, esh,
                                                                         (long long)NEDG * 128);
  }

  // ---- readout ----
  // x1 = relu(concat(h[src],h[dst]) @ W0 + b0) -> bf16 into EB (e mirror dead); W0 panel 64KB LDS
  gemm_gather_bf16<<<gemm_blocks(EDGE_MT), 256, 65536, stream>>>(
      HB, src, dst, PK + PK_W0, mlpb0, EB, EDGE_MT);
  // x2 = relu(x1 @ W1 + b1) -> f32 into B1 (e f32 dead), Nc=64; W1 panel 16KB LDS
  gemm_abf16<4, true><<<gemm_blocks(EDGE_MT), 256, 16384, stream>>>(
      EB, 128, PK + PK_W1, mlpb1, B1, 64, EDGE_MT, 4);
  final_mlp<<<(NEDG + 255) / 256, 256, 0, stream>>>(B1, mlpW2, mlpb2, (float*)d_out);
}
